// RoPEAttentionLayer_9079560864060
// MI455X (gfx1250) — matrix operand order repaired and checked by reading, not compiled, not run on hardware
//
#include <hip/hip_runtime.h>

typedef __attribute__((ext_vector_type(16))) _Float16 v16h;
typedef __attribute__((ext_vector_type(8)))  _Float16 v8h;
typedef __attribute__((ext_vector_type(8)))  float    v8f;
typedef __attribute__((ext_vector_type(4)))  unsigned int u32x4;
typedef __attribute__((ext_vector_type(8)))  int i32x8;
typedef __attribute__((ext_vector_type(4)))  int i32x4;

union V16U { v16h v; v8h h[2]; };

static __device__ __forceinline__ v8f wmma_f16(v16h a, v16h b, v8f c) {
  return __builtin_amdgcn_wmma_f32_16x16x32_f16(false, a, false, b, (short)0, c, false, false);
}

#if __has_builtin(__builtin_amdgcn_tensor_load_to_lds)
#define USE_TDM 1
#else
#define USE_TDM 0
#endif

#if USE_TDM
// Issue one TDM 2D tile load: 64 k-elements x 64 rows of f16 from Bt[n][k]
// (row stride K elements) into contiguous LDS at lds_off.
static __device__ __forceinline__ void tdm_load_tile(const _Float16* gsrc,
                                                     unsigned lds_off,
                                                     int K, int N) {
  unsigned long long ga = (unsigned long long)(uintptr_t)gsrc;
  u32x4 g0;
  g0[0] = 1u;                                           // count=1, user mode
  g0[1] = lds_off;                                      // lds_addr (bytes)
  g0[2] = (unsigned)ga;                                 // global_addr[31:0]
  g0[3] = ((unsigned)(ga >> 32) & 0x01FFFFFFu)          // global_addr[56:32]
          | 0x80000000u;                                // type=2 ("image")
  i32x8 g1;
  g1[0] = (int)(1u << 16);                              // data_size=1 -> 2 bytes
  g1[1] = (int)(((unsigned)K & 0xFFFFu) << 16);         // tensor_dim0[15:0]
  g1[2] = (int)((((unsigned)K >> 16) & 0xFFFFu) |       // tensor_dim0[31:16]
                (((unsigned)N & 0xFFFFu) << 16));       // tensor_dim1[15:0]
  g1[3] = (int)((((unsigned)N >> 16) & 0xFFFFu) |       // tensor_dim1[31:16]
                (64u << 16));                           // tile_dim0=64
  g1[4] = 64;                                           // tile_dim1=64, tile_dim2=0
  g1[5] = K;                                            // tensor_dim0_stride[31:0]
  g1[6] = 0;                                            // stride hi / dim1_stride lo
  g1[7] = 0;
  i32x4 gz4 = {0, 0, 0, 0};                             // groups 2/3: dims 3/4 unused
  i32x8 gz8 = {0, 0, 0, 0, 0, 0, 0, 0};
  __builtin_amdgcn_tensor_load_to_lds(g0, g1, gz4, gz4, gz8, 0);
}
#endif

// ---------------- weight convert + transpose: W[K][N] (row-major) -> Wt[N][K] f16
__global__ void k_transpose_f16(const float* __restrict__ W, _Float16* __restrict__ Wt,
                                int K, int N) {
  long long idx = (long long)blockIdx.x * blockDim.x + threadIdx.x;
  long long total = (long long)K * N;
  if (idx >= total) return;
  int k = (int)(idx / N), n = (int)(idx % N);
  Wt[(long long)n * K + k] = (_Float16)W[idx];
}

// ---------------- LayerNorm -> f16 (one block per row)
__global__ void k_layernorm_f16(const float* __restrict__ X, const float* __restrict__ g,
                                const float* __restrict__ b, _Float16* __restrict__ out,
                                int D) {
  int row = blockIdx.x, t = threadIdx.x;
  long long base = (long long)row * D;
  __shared__ float s1[256], s2[256];
  float a0 = 0.f, a1 = 0.f;
  for (int i = t; i < D; i += 256) { float v = X[base + i]; a0 += v; a1 += v * v; }
  s1[t] = a0; s2[t] = a1; __syncthreads();
  for (int o = 128; o > 0; o >>= 1) {
    if (t < o) { s1[t] += s1[t + o]; s2[t] += s2[t + o]; }
    __syncthreads();
  }
  float mu = s1[0] / D;
  float var = s2[0] / D - mu * mu;
  float rstd = rsqrtf(var + 1e-5f);
  for (int i = t; i < D; i += 256)
    out[base + i] = (_Float16)((X[base + i] - mu) * rstd * g[i] + b[i]);
}

// ---------------- residual add + LayerNorm -> f16 (also stores fp32 residual X2)
__global__ void k_resid_ln_f16(const float* __restrict__ X, const float* __restrict__ Y,
                               const float* __restrict__ g, const float* __restrict__ b,
                               float* __restrict__ X2, _Float16* __restrict__ out, int D) {
  int row = blockIdx.x, t = threadIdx.x;
  long long base = (long long)row * D;
  __shared__ float s1[256], s2[256];
  float a0 = 0.f, a1 = 0.f;
  for (int i = t; i < D; i += 256) {
    float v = X[base + i] + Y[base + i];
    X2[base + i] = v; a0 += v; a1 += v * v;
  }
  s1[t] = a0; s2[t] = a1; __syncthreads();
  for (int o = 128; o > 0; o >>= 1) {
    if (t < o) { s1[t] += s1[t + o]; s2[t] += s2[t + o]; }
    __syncthreads();
  }
  float mu = s1[0] / D;
  float var = s2[0] / D - mu * mu;
  float rstd = rsqrtf(var + 1e-5f);
  for (int i = t; i < D; i += 256)
    out[base + i] = (_Float16)((X2[base + i] - mu) * rstd * g[i] + b[i]);
}

// ---------------- WMMA GEMM with TDM-staged B tiles:
// C[M][N] = A[M][K](f16) * Bt[N][K](f16)^T + bias
// block = 8 waves = 128 rows x 64 cols; B 64x64 f16 tile double-buffered in LDS via TDM
__global__ void __launch_bounds__(256)
k_gemm_f16(const _Float16* __restrict__ A, const _Float16* __restrict__ Bt,
           const float* __restrict__ bias, float* __restrict__ C,
           int M, int N, int K) {
  __shared__ _Float16 Blds[2][64 * 64];
  int wv = threadIdx.x >> 5;
  int lane = threadIdx.x & 31;
  int nW = N >> 6;
  int mi = blockIdx.x / nW, nj = blockIdx.x % nW;
  int nbase = nj << 6;
  int mbase = (mi << 7) + (wv << 4);
  int l = lane & 15, half = lane >> 4;
  int nk = K >> 6;

  v8f z = {0.f, 0.f, 0.f, 0.f, 0.f, 0.f, 0.f, 0.f};
  v8f acc[4] = {z, z, z, z};
  const _Float16* arow = A + (long long)(mbase + l) * K;
  const _Float16* bbase = Bt + (long long)nbase * K;

#if USE_TDM
  unsigned lds0 = (unsigned)(uintptr_t)(&Blds[0][0]);
  unsigned lds1 = (unsigned)(uintptr_t)(&Blds[1][0]);
  if (wv == 0) tdm_load_tile(bbase, lds0, K, N);
#endif

  for (int kc = 0; kc < nk; ++kc) {
    int buf = kc & 1;
#if USE_TDM
    if (wv == 0) {
      if (kc + 1 < nk) {
        tdm_load_tile(bbase + (kc + 1) * 64, (kc & 1) ? lds0 : lds1, K, N);
        __builtin_amdgcn_s_wait_tensorcnt((short)1);
      } else {
        __builtin_amdgcn_s_wait_tensorcnt((short)0);
      }
    }
    __syncthreads();
#else
    __syncthreads();
    {  // cooperative stage: 256 threads x 16 halves = 64x64 tile
      int t = threadIdx.x;
      int rowl = t >> 2;
      int kseg = (t & 3) * 16;
      buf = 0;
      *(v16h*)(&Blds[0][rowl * 64 + kseg]) =
          *(const v16h*)(bbase + (long long)rowl * K + kc * 64 + kseg);
    }
    __syncthreads();
#endif

#pragma unroll
    for (int k0l = 0; k0l < 64; k0l += 32) {
      V16U a;
      a.h[0] = *(const v8h*)(arow + kc * 64 + k0l + half * 16);
      a.h[1] = *(const v8h*)(arow + kc * 64 + k0l + 8 + half * 16);
#pragma unroll
      for (int nt = 0; nt < 4; ++nt) {
        v16h bf = *(const v16h*)(&Blds[buf][(nt * 16 + l) * 64 + k0l + half * 16]);
        acc[nt] = wmma_f16(a.v, bf, acc[nt]);
      }
    }
    __syncthreads();
  }

#pragma unroll
  for (int nt = 0; nt < 4; ++nt) {
#pragma unroll
    for (int r = 0; r < 8; ++r) {
      int m = mbase + half * 8 + r;
      int n = nbase + nt * 16 + l;
      C[(long long)m * N + n] = acc[nt][r] + bias[n];
    }
  }
}

// ---------------- RoPE + repack: qkv fp32 [tok][3*1024] -> Q,K f16 [bh][s][64], Vt f16 [bh][64][s]
__global__ void k_rope_pack(const float* __restrict__ QKV, _Float16* __restrict__ Q,
                            _Float16* __restrict__ Kf, _Float16* __restrict__ Vt,
                            int S, int H) {
  const int hd = 64;
  long long gid = (long long)blockIdx.x * blockDim.x + threadIdx.x;
  int d = (int)(gid & 63);
  long long t1 = gid >> 6;
  int s = (int)(t1 % S);
  long long t2 = t1 / S;
  int h = (int)(t2 % H);
  int b = (int)(t2 / H);
  const float* row = QKV + (long long)(b * S + s) * (3 * H * hd);
  float q = row[h * hd + d];
  float k = row[H * hd + h * hd + d];
  float v = row[2 * H * hd + h * hd + d];
  int dro = (d < 32) ? (d + 32) : (d - 32);
  float qr = row[h * hd + dro];
  float kr = row[H * hd + h * hd + dro];
  float qrot = (d < 32) ? -qr : qr;
  float krot = (d < 32) ? -kr : kr;
  int i2 = d & 31;
  float inv = __expf(-((float)(2 * i2) / 64.0f) * 9.210340371976184f); // 10000^{-2i/64}
  float fr = (float)s * inv;
  float cs = cosf(fr), sn = sinf(fr);
  float qo = q * cs + qrot * sn;
  float ko = k * cs + krot * sn;
  long long bh = (long long)b * H + h;
  Q [ (bh * S + s) * hd + d ] = (_Float16)qo;
  Kf[ (bh * S + s) * hd + d ] = (_Float16)ko;
  Vt[ (bh * hd + d) * S + s ] = (_Float16)v;
}

// ---------------- flash attention: one wave per 16-row q tile, 32-key tiles, WMMA
__global__ void __launch_bounds__(256)
k_attn(const _Float16* __restrict__ Q, const _Float16* __restrict__ Km,
       const _Float16* __restrict__ Vt, _Float16* __restrict__ Out,
       int S, int H) {
  const int hd = 64;
  int wid  = (int)((blockIdx.x * blockDim.x + threadIdx.x) >> 5);
  int lane = threadIdx.x & 31;
  int wb   = threadIdx.x >> 5;
  int qtiles = S >> 4;
  int bh = wid / qtiles, qi = wid % qtiles;
  if (bh >= 32) return;
  int qbase = qi << 4;
  int l = lane & 15, half = lane >> 4;

  const _Float16* Qb = Q  + (long long)bh * S * hd;
  const _Float16* Kb = Km + (long long)bh * S * hd;
  const _Float16* Vb = Vt + (long long)bh * hd * S;

  __shared__ _Float16 Plds[8][16 * 32];

  const _Float16* qrow = Qb + (long long)(qbase + l) * hd;
  V16U qf0, qf1;
  qf0.h[0] = *(const v8h*)(qrow + 0  + half * 16);
  qf0.h[1] = *(const v8h*)(qrow + 8  + half * 16);
  qf1.h[0] = *(const v8h*)(qrow + 32 + half * 16);
  qf1.h[1] = *(const v8h*)(qrow + 40 + half * 16);

  v8f z = {0.f, 0.f, 0.f, 0.f, 0.f, 0.f, 0.f, 0.f};
  v8f o[4] = {z, z, z, z};
  float mprev[8], lprev[8];
#pragma unroll
  for (int r = 0; r < 8; ++r) { mprev[r] = -3.0e38f; lprev[r] = 0.f; }

  int rowg_base = qbase + half * 8;

  for (int j0 = 0; j0 <= qbase; j0 += 32) {
    v8f s0 = z, s1 = z;
    {
      v16h b00 = *(const v16h*)(Kb + (long long)(j0 + l) * hd + 0  + half * 16);
      v16h b01 = *(const v16h*)(Kb + (long long)(j0 + l) * hd + 32 + half * 16);
      s0 = wmma_f16(qf0.v, b00, s0);
      s0 = wmma_f16(qf1.v, b01, s0);
      v16h b10 = *(const v16h*)(Kb + (long long)(j0 + 16 + l) * hd + 0  + half * 16);
      v16h b11 = *(const v16h*)(Kb + (long long)(j0 + 16 + l) * hd + 32 + half * 16);
      s1 = wmma_f16(qf0.v, b10, s1);
      s1 = wmma_f16(qf1.v, b11, s1);
    }
    float p0[8], p1[8], alpha[8];
#pragma unroll
    for (int r = 0; r < 8; ++r) {
      float v0 = s0[r] * 0.125f;
      float v1 = s1[r] * 0.125f;
      if (j0 + l      > rowg_base + r) v0 = -3.0e38f;
      if (j0 + 16 + l > rowg_base + r) v1 = -3.0e38f;
      float t = fmaxf(v0, v1);
      t = fmaxf(t, __shfl_xor(t, 1));
      t = fmaxf(t, __shfl_xor(t, 2));
      t = fmaxf(t, __shfl_xor(t, 4));
      t = fmaxf(t, __shfl_xor(t, 8));
      float mn = fmaxf(mprev[r], t);
      alpha[r] = __expf(mprev[r] - mn);
      p0[r] = __expf(v0 - mn);
      p1[r] = __expf(v1 - mn);
      float sm = p0[r] + p1[r];
      sm += __shfl_xor(sm, 1);
      sm += __shfl_xor(sm, 2);
      sm += __shfl_xor(sm, 4);
      sm += __shfl_xor(sm, 8);
      lprev[r] = alpha[r] * lprev[r] + sm;
      mprev[r] = mn;
    }
#pragma unroll
    for (int nt = 0; nt < 4; ++nt)
#pragma unroll
      for (int r = 0; r < 8; ++r) o[nt][r] *= alpha[r];

#pragma unroll
    for (int r = 0; r < 8; ++r) {
      int row = half * 8 + r;
      Plds[wb][row * 32 + l]      = (_Float16)p0[r];
      Plds[wb][row * 32 + 16 + l] = (_Float16)p1[r];
    }
    asm volatile("s_wait_dscnt 0x0" ::: "memory");
    V16U pf;
    pf.h[0] = *(const v8h*)(&Plds[wb][l * 32 + half * 16]);
    pf.h[1] = *(const v8h*)(&Plds[wb][l * 32 + 8 + half * 16]);

#pragma unroll
    for (int nt = 0; nt < 4; ++nt) {
      v16h vf = *(const v16h*)(Vb + (long long)(nt * 16 + l) * S + j0 + half * 16);
      o[nt] = wmma_f16(pf.v, vf, o[nt]);
    }
  }

  int bb = bh / H, hh = bh % H;
#pragma unroll
  for (int r = 0; r < 8; ++r) {
    float invl = 1.0f / lprev[r];
    int srow = qbase + half * 8 + r;
    long long base = ((long long)bb * S + srow) * (H * hd) + hh * hd;
#pragma unroll
    for (int nt = 0; nt < 4; ++nt)
      Out[base + nt * 16 + l] = (_Float16)(o[nt][r] * invl);
  }
}

// ---------------- exact GELU fp32 -> f16
__global__ void k_gelu_f16(const float* __restrict__ in, _Float16* __restrict__ out,
                           long long n) {
  long long i = (long long)blockIdx.x * blockDim.x + threadIdx.x;
  if (i >= n) return;
  float v = in[i];
  out[i] = (_Float16)(0.5f * v * (1.0f + erff(v * 0.70710678118654752f)));
}

// ---------------- final residual add
__global__ void k_add(const float* __restrict__ a, const float* __restrict__ b,
                      float* __restrict__ o, long long n) {
  long long i = (long long)blockIdx.x * blockDim.x + threadIdx.x;
  if (i >= n) return;
  o[i] = a[i] + b[i];
}

extern "C" void kernel_launch(void* const* d_in, const int* in_sizes, int n_in,
                              void* d_out, int out_size, void* d_ws, size_t ws_size,
                              hipStream_t stream) {
  const float* x    = (const float*)d_in[0];
  const float* ln1g = (const float*)d_in[1];
  const float* ln1b = (const float*)d_in[2];
  const float* Wqkv = (const float*)d_in[3];
  const float* bqkv = (const float*)d_in[4];
  const float* Wo   = (const float*)d_in[5];
  const float* bo   = (const float*)d_in[6];
  const float* ln2g = (const float*)d_in[7];
  const float* ln2b = (const float*)d_in[8];
  const float* W1   = (const float*)d_in[9];
  const float* b1   = (const float*)d_in[10];
  const float* W2   = (const float*)d_in[11];
  const float* b2   = (const float*)d_in[12];
  float* out = (float*)d_out;
  char* wsb = (char*)d_ws;

  const int Bb = 2, S = 2048, D = 1024, H = 16, HD = 64;
  const int M = Bb * S; // 4096 tokens

  // workspace layout (bytes), regions reused across pipeline phases
  _Float16* WqkvT = (_Float16*)(wsb + 0ll);                        // 6 MB
  _Float16* WoT   = (_Float16*)(wsb + 6291456ll);                  // 2 MB
  _Float16* W1T   = (_Float16*)(wsb + 8388608ll);                  // 8 MB
  _Float16* W2T   = (_Float16*)(wsb + 16777216ll);                 // 8 MB
  _Float16* XN    = (_Float16*)(wsb + 25165824ll);                 // 8 MB (LN1 out, later LN2 out)
  float*    BIGF  = (float*)   (wsb + 33554432ll);                 // 64 MB (qkv f32, later mlp1 f32)
  _Float16* Qf    = (_Float16*)(wsb + 100663296ll);                // 8 MB
  _Float16* Kf    = (_Float16*)(wsb + 100663296ll + 8388608ll);    // 8 MB
  _Float16* Vtf   = (_Float16*)(wsb + 100663296ll + 16777216ll);   // 8 MB
  _Float16* ATTN  = (_Float16*)(wsb + 100663296ll + 25165824ll);   // 8 MB
  _Float16* Gf    = (_Float16*)(wsb + 100663296ll);                // 32 MB (reuses Q/K/Vt/ATTN)
  float*    PROJ  = (float*)   (wsb + 134217728ll);                // 16 MB (Wo out, later mlp2 out)
  float*    X2    = (float*)   (wsb + 150994944ll);                // 16 MB residual

  dim3 blk(256);

  // 1. weight f16 transposes
  k_transpose_f16<<<(D * 3 * D + 255) / 256, blk, 0, stream>>>(Wqkv, WqkvT, D, 3 * D);
  k_transpose_f16<<<(D * D + 255) / 256,     blk, 0, stream>>>(Wo,   WoT,   D, D);
  k_transpose_f16<<<(D * 4 * D + 255) / 256, blk, 0, stream>>>(W1,   W1T,   D, 4 * D);
  k_transpose_f16<<<(4 * D * D + 255) / 256, blk, 0, stream>>>(W2,   W2T,   4 * D, D);
  // 2. LN1
  k_layernorm_f16<<<M, blk, 0, stream>>>(x, ln1g, ln1b, XN, D);
  // 3. QKV GEMM (4096 x 3072 x 1024), 128x64 block tiles
  k_gemm_f16<<<(M / 128) * (3 * D / 64), blk, 0, stream>>>(XN, WqkvT, bqkv, BIGF, M, 3 * D, D);
  // 4. RoPE + repack
  k_rope_pack<<<(Bb * H * S * HD) / 256, blk, 0, stream>>>(BIGF, Qf, Kf, Vtf, S, H);
  // 5. causal flash attention
  k_attn<<<(Bb * H * (S / 16)) / 8, blk, 0, stream>>>(Qf, Kf, Vtf, ATTN, S, H);
  // 6. output projection (4096 x 1024 x 1024)
  k_gemm_f16<<<(M / 128) * (D / 64), blk, 0, stream>>>(ATTN, WoT, bo, PROJ, M, D, D);
  // 7. residual + LN2
  k_resid_ln_f16<<<M, blk, 0, stream>>>(x, PROJ, ln2g, ln2b, X2, XN, D);
  // 8. MLP up (4096 x 4096 x 1024)
  k_gemm_f16<<<(M / 128) * (4 * D / 64), blk, 0, stream>>>(XN, W1T, b1, BIGF, M, 4 * D, D);
  // 9. exact GELU -> f16
  k_gelu_f16<<<(int)(((long long)M * 4 * D) / 256), blk, 0, stream>>>(BIGF, Gf, (long long)M * 4 * D);
  // 10. MLP down (4096 x 1024 x 4096)
  k_gemm_f16<<<(M / 128) * (D / 64), blk, 0, stream>>>(Gf, W2T, b2, PROJ, M, D, 4 * D);
  // 11. final residual add
  k_add<<<(M * D) / 256, blk, 0, stream>>>(X2, PROJ, out, (long long)M * D);
}
